// SimpleMoEModel_89867895701965
// MI455X (gfx1250) — matrix operand order repaired and checked by reading, compiled, not run on hardware
//
#include <hip/hip_runtime.h>
#include <hip/hip_bf16.h>
#include <stdint.h>

// Problem constants (match reference)
#define BB 8192
#define DD 1024
#define EE 8
#define HH 4096
#define CC 10

typedef __attribute__((ext_vector_type(16))) __bf16 v16bf;
typedef __attribute__((ext_vector_type(8)))  float  v8f;
typedef int i4v __attribute__((vector_size(16)));  // builtin's 128-bit payload type

#define ASYNC_OK __has_builtin(__builtin_amdgcn_global_load_async_to_lds_b128)

__device__ __forceinline__ unsigned short f2bf(float f) {
  union { float f; uint32_t u; } v; v.f = f;
  uint32_t u = v.u;
  return (unsigned short)((u + 0x7FFFu + ((u >> 16) & 1u)) >> 16); // RNE
}

#if ASYNC_OK
// 16B global -> LDS async copy; the same imm offset applies to both sides (ISA §4.4).
template <int OFFS>
__device__ __forceinline__ void async_b128(const void* g, void* l) {
  __builtin_amdgcn_global_load_async_to_lds_b128(
      (__attribute__((address_space(1))) i4v*)(g),
      (__attribute__((address_space(3))) i4v*)(l), OFFS, 0);
}
#endif

template <int N>
__device__ __forceinline__ void wait_async() {
#if __has_builtin(__builtin_amdgcn_s_wait_asynccnt)
  __builtin_amdgcn_s_wait_asynccnt(N);
#elif defined(__gfx1250__)
  asm volatile("s_wait_asynccnt %0" ::"n"(N) : "memory");
#endif
}

// ---------------- router: routing = softmax(x @ Wr + br) ----------------
__global__ void router_kernel(const float* __restrict__ x, const float* __restrict__ Wr,
                              const float* __restrict__ br, float* __restrict__ routing) {
  int b = blockIdx.x * blockDim.x + threadIdx.x;
  if (b >= BB) return;
  float acc[EE];
  #pragma unroll
  for (int e = 0; e < EE; ++e) acc[e] = br[e];
  const float* xr = x + (size_t)b * DD;
  for (int d = 0; d < DD; ++d) {
    float xv = xr[d];
    const float* w = Wr + d * EE;
    #pragma unroll
    for (int e = 0; e < EE; ++e) acc[e] += xv * w[e];
  }
  float m = acc[0];
  #pragma unroll
  for (int e = 1; e < EE; ++e) m = fmaxf(m, acc[e]);
  float s = 0.f;
  #pragma unroll
  for (int e = 0; e < EE; ++e) { acc[e] = __expf(acc[e] - m); s += acc[e]; }
  float inv = 1.0f / s;
  #pragma unroll
  for (int e = 0; e < EE; ++e) routing[(size_t)b * EE + e] = acc[e] * inv;
}

// ---------------- elementwise f32 -> bf16 ----------------
__global__ void cvt_bf16_kernel(const float* __restrict__ src, unsigned short* __restrict__ dst, int n) {
  int i = blockIdx.x * blockDim.x + threadIdx.x;
  if (i < n) dst[i] = f2bf(src[i]);
}

// src[rows][cols] (f32) -> dst[cols][rows] (bf16)
__global__ void transpose_cvt_kernel(const float* __restrict__ src, unsigned short* __restrict__ dst,
                                     int rows, int cols) {
  int i = blockIdx.x * blockDim.x + threadIdx.x;
  if (i < rows * cols) {
    int r = i / cols, c = i % cols;
    dst[(size_t)c * rows + r] = f2bf(src[i]);
  }
}

__global__ void zero_kernel(float* __restrict__ p, int n) {
  int i = blockIdx.x * blockDim.x + threadIdx.x;
  if (i < n) p[i] = 0.0f;
}

// ---------------- tiled bf16 WMMA GEMM ----------------
// C-tile = A[M,K] x Bt[N,K]^T.  MODE 0: outH = bf16(relu(v + bias[n]))
//                               MODE 1: outAcc[m,n] += routing[m,expert] * (v + bias[n])
// Block: 256 threads (8 waves), 128x128 output tile; wave = 64x32 = 4x2 WMMA tiles.
// Async path: 3-deep LDS pipeline fed by GLOBAL_LOAD_ASYNC_TO_LDS_B128 (ASYNCcnt),
// one workgroup barrier per 32-wide K panel; WMMAs never wait on global latency.
template <int MODE>
__global__ __launch_bounds__(256) void gemm_bf16_kernel(
    const unsigned short* __restrict__ A,   // [M,K] bf16 row-major
    const unsigned short* __restrict__ Bt,  // [N,K] bf16 row-major (i.e. B transposed)
    const float* __restrict__ bias,         // [N]
    const float* __restrict__ routing,      // [M,EE] (MODE 1)
    int expert, int M, int N, int K,
    unsigned short* __restrict__ outH,      // MODE 0: [M,N] bf16
    float* __restrict__ outAcc)             // MODE 1: [M,N] f32 (RMW)
{
#if ASYNC_OK
  __shared__ __align__(16) unsigned short Alds[3][128 * 32];
  __shared__ __align__(16) unsigned short Blds[3][128 * 32];
#else
  __shared__ __align__(16) unsigned short Alds[1][128 * 32];
  __shared__ __align__(16) unsigned short Blds[1][128 * 32];
#endif

  const int t    = threadIdx.x;
  const int lane = t & 31;
  const int w    = t >> 5;        // wave 0..7
  const int wm   = w >> 2;        // 0..1  -> 64-row slab
  const int wn   = w & 3;         // 0..3  -> 32-col slab
  const int m0   = blockIdx.y * 128;
  const int n0   = blockIdx.x * 128;
  const int lrow = t >> 1;        // 0..127 : staging row
  const int lseg = (t & 1) * 16;  // element offset 0/16 within 32-wide K panel
  const int h    = lane >> 4;     // lane half (ISA K-chunk select)
  const int l15  = lane & 15;

  v8f acc[4][2];
  #pragma unroll
  for (int i = 0; i < 4; ++i)
    #pragma unroll
    for (int j = 0; j < 2; ++j) acc[i][j] = v8f{};

  const size_t arow = (size_t)(m0 + lrow) * K + lseg;
  const size_t brow = (size_t)(n0 + lrow) * K + lseg;
  const int ldoff = lrow * 32 + lseg;

  // One 32-wide K panel of WMMA work from LDS buffer `buf`.
  auto compute_panel = [&](const unsigned short* Asrc, const unsigned short* Bsrc) {
    const __bf16* Ab = (const __bf16*)Asrc;
    const __bf16* Bb = (const __bf16*)Bsrc;
    // A fragment (16-bit A 16x32): lane half h holds K = {8h..8h+7} U {16+8h..16+8h+7}
    v16bf afrag[4];
    #pragma unroll
    for (int i = 0; i < 4; ++i) {
      const __bf16* ap = Ab + (wm * 64 + i * 16 + l15) * 32;
      #pragma unroll
      for (int j = 0; j < 8; ++j) {
        afrag[i][j]     = ap[8 * h + j];
        afrag[i][8 + j] = ap[16 + 8 * h + j];
      }
    }
    // B fragment (16-bit B 32x16): lane = N, half h holds K = {8h..8h+7} U {16+8h..16+8h+7}
    v16bf bfrag[2];
    #pragma unroll
    for (int j2 = 0; j2 < 2; ++j2) {
      const __bf16* bp = Bb + (wn * 32 + j2 * 16 + l15) * 32;
      #pragma unroll
      for (int j = 0; j < 8; ++j) {
        bfrag[j2][j]     = bp[8 * h + j];
        bfrag[j2][8 + j] = bp[16 + 8 * h + j];
      }
    }
    #pragma unroll
    for (int i = 0; i < 4; ++i)
      #pragma unroll
      for (int j2 = 0; j2 < 2; ++j2)
        acc[i][j2] = __builtin_amdgcn_wmma_f32_16x16x32_bf16(
            false, afrag[i], false, bfrag[j2], (short)0, acc[i][j2], false, false);
  };

#if ASYNC_OK
  const int nk = K >> 5;
  // Each thread stages 32B of A and 32B of B per tile: 4 async b128 ops / tile.
  auto issue_tile = [&](int kk, int buf) {
    const unsigned short* ga = A + arow + (size_t)kk * 32;
    const unsigned short* gb = Bt + brow + (size_t)kk * 32;
    unsigned short* la = &Alds[buf][ldoff];
    unsigned short* lb = &Blds[buf][ldoff];
    async_b128<0>(ga, la);
    async_b128<16>(ga, la);
    async_b128<0>(gb, lb);
    async_b128<16>(gb, lb);
  };

  issue_tile(0, 0);
  if (nk > 1) issue_tile(1, 1);
  for (int kk = 0; kk < nk; ++kk) {
    // In-order ASYNCcnt: leave the newest tile (4 ops) pending, oldest is done.
    if (kk + 1 < nk) wait_async<4>(); else wait_async<0>();
    __syncthreads();                 // tile kk visible to all waves; also fences
                                     // last iteration's reads of buf[(kk+2)%3]
    if (kk + 2 < nk) issue_tile(kk + 2, (kk + 2) % 3);
    compute_panel(Alds[kk % 3], Blds[kk % 3]);
  }
  __syncthreads();
#else
  // Fallback: synchronous single-buffer staging (known-good).
  for (int kk = 0; kk < K; kk += 32) {
    __syncthreads();
    const uint4* ga = (const uint4*)(A + arow + kk);
    uint4* la = (uint4*)(&Alds[0][ldoff]);
    la[0] = ga[0]; la[1] = ga[1];
    const uint4* gb = (const uint4*)(Bt + brow + kk);
    uint4* lb = (uint4*)(&Blds[0][ldoff]);
    lb[0] = gb[0]; lb[1] = gb[1];
    __syncthreads();
    compute_panel(Alds[0], Blds[0]);
  }
#endif

  // Epilogue. C/D layout: element r -> m = r + 8h, n = lane&15.
  #pragma unroll
  for (int i = 0; i < 4; ++i) {
    #pragma unroll
    for (int j2 = 0; j2 < 2; ++j2) {
      const int gn = n0 + wn * 32 + j2 * 16 + l15;
      const float bv = bias[gn];
      #pragma unroll
      for (int r = 0; r < 8; ++r) {
        const int gm = m0 + wm * 64 + i * 16 + r + 8 * h;
        const float v = acc[i][j2][r] + bv;
        if (MODE == 0) {
          outH[(size_t)gm * N + gn] = f2bf(fmaxf(v, 0.0f));
        } else {
          const float wgt = routing[(size_t)gm * EE + expert];
          outAcc[(size_t)gm * N + gn] += wgt * v;  // race-free: experts sequential
        }
      }
    }
  }
}

// ---------------- classifier: out = acc @ Wc + bc ----------------
__global__ void classifier_kernel(const float* __restrict__ acc, const float* __restrict__ Wc,
                                  const float* __restrict__ bc, float* __restrict__ out) {
  int t = blockIdx.x * blockDim.x + threadIdx.x;
  if (t >= BB * CC) return;
  int b = t / CC, c = t % CC;
  float s = bc[c];
  const float* ar = acc + (size_t)b * DD;
  for (int d = 0; d < DD; ++d) s += ar[d] * Wc[d * CC + c];
  out[t] = s;
}

// ---------------- launch ----------------
extern "C" void kernel_launch(void* const* d_in, const int* in_sizes, int n_in,
                              void* d_out, int out_size, void* d_ws, size_t ws_size,
                              hipStream_t stream) {
  const float* x  = (const float*)d_in[0];
  const float* Wr = (const float*)d_in[1];
  const float* br = (const float*)d_in[2];
  const float* W1 = (const float*)d_in[3];
  const float* b1 = (const float*)d_in[4];
  const float* W2 = (const float*)d_in[5];
  const float* b2 = (const float*)d_in[6];
  const float* Wc = (const float*)d_in[7];
  const float* bc = (const float*)d_in[8];
  float* out = (float*)d_out;

  // Workspace carve-up (256B aligned)
  uint8_t* ws = (uint8_t*)d_ws;
  size_t off = 0;
  auto carve = [&](size_t bytes) { size_t o = off; off += (bytes + 255) & ~(size_t)255; return o; };
  float*          routing = (float*)         (ws + carve((size_t)BB * EE * 4));
  unsigned short* xh      = (unsigned short*)(ws + carve((size_t)BB * DD * 2));
  unsigned short* w1t     = (unsigned short*)(ws + carve((size_t)DD * HH * 2));
  unsigned short* w2t     = (unsigned short*)(ws + carve((size_t)HH * DD * 2));
  unsigned short* hbuf    = (unsigned short*)(ws + carve((size_t)BB * HH * 2));
  float*          accb    = (float*)         (ws + carve((size_t)BB * DD * 4));
  (void)ws_size; (void)in_sizes; (void)n_in; (void)out_size;

  // 1) routing (f32, exact)
  router_kernel<<<(BB + 255) / 256, 256, 0, stream>>>(x, Wr, br, routing);
  // 2) x -> bf16
  cvt_bf16_kernel<<<(BB * DD + 255) / 256, 256, 0, stream>>>(x, xh, BB * DD);
  // 3) acc = 0
  zero_kernel<<<(BB * DD + 255) / 256, 256, 0, stream>>>(accb, BB * DD);

  // 4) per-expert fused pipeline (stream-ordered => sequential, race-free)
  for (int e = 0; e < EE; ++e) {
    // W1[e] : [D,H] -> w1t [H,D] bf16 ;  W2[e] : [H,D] -> w2t [D,H] bf16
    transpose_cvt_kernel<<<(DD * HH + 255) / 256, 256, 0, stream>>>(
        W1 + (size_t)e * DD * HH, w1t, DD, HH);
    transpose_cvt_kernel<<<(HH * DD + 255) / 256, 256, 0, stream>>>(
        W2 + (size_t)e * HH * DD, w2t, HH, DD);
    // h = relu(x @ W1[e] + b1[e]) : M=B, N=H, K=D
    gemm_bf16_kernel<0><<<dim3(HH / 128, BB / 128), 256, 0, stream>>>(
        xh, w1t, b1 + (size_t)e * HH, nullptr, e, BB, HH, DD, hbuf, nullptr);
    // acc += routing[:,e] * (h @ W2[e] + b2[e]) : M=B, N=D, K=H
    gemm_bf16_kernel<1><<<dim3(DD / 128, BB / 128), 256, 0, stream>>>(
        hbuf, w2t, b2 + (size_t)e * DD, routing, e, BB, DD, HH, nullptr, accb);
  }

  // 5) out = acc @ Wc + bc
  classifier_kernel<<<(BB * CC + 255) / 256, 256, 0, stream>>>(accb, Wc, bc, out);
}
